// MultiheadSupervisedContrastiveLossNoNegatives_12687333392924
// MI455X (gfx1250) — matrix operand order repaired and checked by reading, not compiled, not run on hardware
//
#include <hip/hip_runtime.h>

typedef __attribute__((ext_vector_type(16))) _Float16 v16h;
typedef __attribute__((ext_vector_type(8)))  float    v8f;

#define BSZ   2048
#define NVIEW 2
#define EDIM  256
#define NH    4
#define FS    64
#define NN    4096   // BSZ*NVIEW
#define NTI   256    // NN/16
#define INV_T 14.2857142857142857f  // 1/0.07

// ---------------------------------------------------------------------------
// prep: normalize features and scatter into WMMA A-fragment / B-fragment
// layouts (ISA 7.12.2, 16-bit A 16x32 and B 32x16, wave32).
// f[v*B + b] = features[b, v, :]  (torch.cat(unbind(dim=1)))
// ---------------------------------------------------------------------------
__global__ void prep_kernel(const float* __restrict__ feat,
                            _Float16* __restrict__ Abuf,
                            _Float16* __restrict__ Bbuf) {
  int n = blockIdx.x;      // 0..4095 row of f
  int e = threadIdx.x;     // 0..255 element within row
  int b = n & (BSZ - 1);
  int v = n >> 11;
  float x = feat[(size_t)b * (NVIEW * EDIM) + v * EDIM + e];
  int h = e >> 6;
  int k = e & 63;

  __shared__ float red[EDIM];
  red[e] = x * x;
  __syncthreads();
  for (int s = 32; s > 0; s >>= 1) {
    if (k < s) red[e] += red[e + s];
    __syncthreads();
  }
  float denom = fmaxf(sqrtf(red[h * 64]), 1e-12f);
  _Float16 hv = (_Float16)(x / denom);

  int kc  = k >> 5;     // which K-chunk of 32
  int K   = k & 31;
  int odd = K & 1;
  int Kp  = K - odd;

  // A-matrix slot: lanes 0-15 hold K {0..7,16..23}, lanes 16-31 {8..15,24..31}
  {
    int ti   = n >> 4;
    int m    = n & 15;
    int hla  = (Kp >> 3) & 1;
    int blk  = Kp >> 4;
    int vv   = ((Kp & 7) >> 1) + 4 * blk;
    int lane = m + 16 * hla;
    int t    = 2 * vv + odd;
    size_t idx = ((((size_t)h * NTI + ti) * 2 + kc) * 32 + lane) * 16 + t;
    Abuf[idx] = hv;
  }
  // B-matrix slot: lanes 0-15 hold K {0..7,16..23}, lanes 16-31 {8..15,24..31}
  {
    int tj   = n >> 4;
    int col  = n & 15;
    int hlb  = (Kp >> 3) & 1;
    int vv   = ((Kp & 7) >> 1) + 4 * (Kp >> 4);
    int lane = col + 16 * hlb;
    int t    = 2 * vv + odd;
    size_t idx = ((((size_t)h * NTI + tj) * 2 + kc) * 32 + lane) * 16 + t;
    Bbuf[idx] = hv;
  }
}

// ---------------------------------------------------------------------------
// mask: bitmask[bi][bj] = (labels[bi] . labels[bj]) > 0, packed 32 bits/word
// ---------------------------------------------------------------------------
__global__ void mask_kernel(const float* __restrict__ labels,
                            unsigned* __restrict__ maskbits) {
  int idx = blockIdx.x * blockDim.x + threadIdx.x;
  if (idx >= BSZ * (BSZ / 32)) return;
  int bi = idx >> 6;
  int wd = idx & 63;
  float li[10];
#pragma unroll
  for (int t = 0; t < 10; ++t) li[t] = labels[bi * 10 + t];
  unsigned bits = 0;
  for (int bit = 0; bit < 32; ++bit) {
    int bj = wd * 32 + bit;
    float d = 0.f;
#pragma unroll
    for (int t = 0; t < 10; ++t) d += li[t] * labels[bj * 10 + t];
    bits |= (d > 0.f ? 1u : 0u) << bit;
  }
  maskbits[idx] = bits;
}

__global__ void zero_kernel(float* accum) { accum[0] = 0.f; accum[1] = 0.f; }
__global__ void fin_kernel(const float* accum, float* out) {
  out[0] = accum[0] / accum[1];
}

// ---------------------------------------------------------------------------
// main fused kernel: 1 block = one 16-row M tile; 4 waves split the j tiles.
// Pass 1: WMMA + per-(row,head) max. Pass 2: recompute WMMA tiles, exp-sums,
// head argmax, masked numerator/counters. Finite-filtered atomic mean.
// ---------------------------------------------------------------------------
__global__ __launch_bounds__(128) void scl_main_kernel(
    const _Float16* __restrict__ Abuf,
    const _Float16* __restrict__ Bbuf,
    const unsigned* __restrict__ maskbits,
    float* __restrict__ accum) {
  const int iti  = blockIdx.x;          // 0..255
  const int lane = threadIdx.x & 31;
  const int w    = threadIdx.x >> 5;    // wave id 0..3
  const int hl   = lane >> 4;           // half-wave
  const int m    = lane & 15;

  __shared__ float lds_m[4][16][NH];
  __shared__ float lds_S[4][16][NH];
  __shared__ float lds_C[4][16][NH];
  __shared__ float lds_A[4][16];

  // A fragments for this block's 16 rows: 4 heads x 2 K-chunks
  v16h a[NH][2];
#pragma unroll
  for (int h = 0; h < NH; ++h)
#pragma unroll
    for (int kc = 0; kc < 2; ++kc)
      a[h][kc] = *(const v16h*)(Abuf +
                   ((((size_t)h * NTI + iti) * 2 + kc) * 32 + lane) * 16);

  float mx[NH][8];
#pragma unroll
  for (int h = 0; h < NH; ++h)
#pragma unroll
    for (int r = 0; r < 8; ++r) mx[h][r] = -3.0e38f;

  // ----- pass 1: running max over all j (includes j==i, as in reference) ---
  for (int tj = w; tj < NTI; tj += 4) {
    if (tj + 4 < NTI)
      __builtin_prefetch(
          (const void*)(Bbuf + ((((size_t)(tj + 4)) * 2) * 32 + lane) * 16), 0, 1);
#pragma unroll
    for (int h = 0; h < NH; ++h) {
      const _Float16* bp =
          Bbuf + ((((size_t)h * NTI + tj) * 2) * 32 + lane) * 16;
      v16h b0 = *(const v16h*)bp;
      v16h b1 = *(const v16h*)(bp + 512);
      v8f acc = {};
      acc = __builtin_amdgcn_wmma_f32_16x16x32_f16(false, a[h][0], false, b0,
                                                   (short)0, acc, false, false);
      acc = __builtin_amdgcn_wmma_f32_16x16x32_f16(false, a[h][1], false, b1,
                                                   (short)0, acc, false, false);
#pragma unroll
      for (int r = 0; r < 8; ++r) mx[h][r] = fmaxf(mx[h][r], acc[r]);
    }
  }

  // reduce max across the 16 lanes of each half-wave (columns of the D tile)
#pragma unroll
  for (int h = 0; h < NH; ++h)
#pragma unroll
    for (int r = 0; r < 8; ++r) {
      float vv = mx[h][r];
#pragma unroll
      for (int s = 1; s < 16; s <<= 1) vv = fmaxf(vv, __shfl_xor(vv, s, 32));
      mx[h][r] = vv;
    }
  if (m == 0) {
#pragma unroll
    for (int r = 0; r < 8; ++r)
#pragma unroll
      for (int h = 0; h < NH; ++h) lds_m[w][r + 8 * hl][h] = mx[h][r];
  }
  __syncthreads();

  float maxT[8][NH];  // maxh * (1/T) for the rows this lane holds
#pragma unroll
  for (int r = 0; r < 8; ++r)
#pragma unroll
    for (int h = 0; h < NH; ++h) {
      int row = r + 8 * hl;
      float mv = fmaxf(fmaxf(lds_m[0][row][h], lds_m[1][row][h]),
                       fmaxf(lds_m[2][row][h], lds_m[3][row][h]));
      maxT[r][h] = mv * INV_T;
    }

  // ----- pass 2 -----
  float S[8][NH], C[8][NH], Aacc[8];
#pragma unroll
  for (int r = 0; r < 8; ++r) {
    Aacc[r] = 0.f;
#pragma unroll
    for (int h = 0; h < NH; ++h) { S[r][h] = 0.f; C[r][h] = 0.f; }
  }

  for (int tj = w; tj < NTI; tj += 4) {
    if (tj + 4 < NTI)
      __builtin_prefetch(
          (const void*)(Bbuf + ((((size_t)(tj + 4)) * 2) * 32 + lane) * 16), 0, 1);
    v8f acc[NH];
#pragma unroll
    for (int h = 0; h < NH; ++h) {
      const _Float16* bp =
          Bbuf + ((((size_t)h * NTI + tj) * 2) * 32 + lane) * 16;
      v16h b0 = *(const v16h*)bp;
      v16h b1 = *(const v16h*)(bp + 512);
      v8f c = {};
      c = __builtin_amdgcn_wmma_f32_16x16x32_f16(false, a[h][0], false, b0,
                                                 (short)0, c, false, false);
      c = __builtin_amdgcn_wmma_f32_16x16x32_f16(false, a[h][1], false, b1,
                                                 (short)0, c, false, false);
      acc[h] = c;
    }
    int j     = tj * 16 + m;        // this lane's column
    int bj    = j & (BSZ - 1);
    int wword = bj >> 5;
    int wbit  = bj & 31;
#pragma unroll
    for (int r = 0; r < 8; ++r) {
      int irow = iti * 16 + r + 8 * hl;
      int bi   = irow & (BSZ - 1);
      unsigned mb = maskbits[bi * 64 + wword];
      float lm = (irow != j) ? 1.f : 0.f;              // logits_mask
      float pm = ((mb >> wbit) & 1u) ? lm : 0.f;       // positive mask
      float d0 = acc[0][r] * INV_T;
      float d1 = acc[1][r] * INV_T;
      float d2 = acc[2][r] * INV_T;
      float d3 = acc[3][r] * INV_T;
      float e0 = __expf(d0 - maxT[r][0]);
      float e1 = __expf(d1 - maxT[r][1]);
      float e2 = __expf(d2 - maxT[r][2]);
      float e3 = __expf(d3 - maxT[r][3]);
      S[r][0] += lm * e0; S[r][1] += lm * e1;
      S[r][2] += lm * e2; S[r][3] += lm * e3;
      // argmax over heads, first max wins (jnp.argmax semantics)
      float dm = d0; int hs = 0;
      if (d1 > dm) { dm = d1; hs = 1; }
      if (d2 > dm) { dm = d2; hs = 2; }
      if (d3 > dm) { dm = d3; hs = 3; }
      float s0 = (hs == 0) ? pm : 0.f;
      float s1 = (hs == 1) ? pm : 0.f;
      float s2 = (hs == 2) ? pm : 0.f;
      float s3 = (hs == 3) ? pm : 0.f;
      Aacc[r] += s0 * (d0 - maxT[r][0]) + s1 * (d1 - maxT[r][1]) +
                 s2 * (d2 - maxT[r][2]) + s3 * (d3 - maxT[r][3]);
      C[r][0] += s0; C[r][1] += s1; C[r][2] += s2; C[r][3] += s3;
    }
  }

  // sum across the 16 lanes of each half
#pragma unroll
  for (int r = 0; r < 8; ++r) {
#pragma unroll
    for (int h = 0; h < NH; ++h) {
      float v1 = S[r][h], v2 = C[r][h];
#pragma unroll
      for (int s = 1; s < 16; s <<= 1) {
        v1 += __shfl_xor(v1, s, 32);
        v2 += __shfl_xor(v2, s, 32);
      }
      S[r][h] = v1; C[r][h] = v2;
    }
    float v3 = Aacc[r];
#pragma unroll
    for (int s = 1; s < 16; s <<= 1) v3 += __shfl_xor(v3, s, 32);
    Aacc[r] = v3;
  }
  if (m == 0) {
#pragma unroll
    for (int r = 0; r < 8; ++r) {
      int row = r + 8 * hl;
#pragma unroll
      for (int h = 0; h < NH; ++h) {
        lds_S[w][row][h] = S[r][h];
        lds_C[w][row][h] = C[r][h];
      }
      lds_A[w][row] = Aacc[r];
    }
  }
  __syncthreads();

  // wave 0: lanes 0 and 16 finalize 8 rows each
  if (w == 0 && m == 0) {
    float lsum = 0.f, lcnt = 0.f;
#pragma unroll
    for (int r = 0; r < 8; ++r) {
      int row = r + 8 * hl;
      float num = lds_A[0][row] + lds_A[1][row] + lds_A[2][row] + lds_A[3][row];
      float Pn = 0.f;
#pragma unroll
      for (int h = 0; h < NH; ++h) {
        float Sh = lds_S[0][row][h] + lds_S[1][row][h] +
                   lds_S[2][row][h] + lds_S[3][row][h];
        float Ch = lds_C[0][row][h] + lds_C[1][row][h] +
                   lds_C[2][row][h] + lds_C[3][row][h];
        num -= Ch * __logf(Sh);
        Pn  += Ch;
      }
      float loss = -(num / Pn);   // TEMPERATURE / BASE_TEMPERATURE == 1
      if (__builtin_isfinite(loss)) { lsum += loss; lcnt += 1.f; }
    }
    atomicAdd(&accum[0], lsum);
    atomicAdd(&accum[1], lcnt);
  }
}

// ---------------------------------------------------------------------------
extern "C" void kernel_launch(void* const* d_in, const int* in_sizes, int n_in,
                              void* d_out, int out_size, void* d_ws,
                              size_t ws_size, hipStream_t stream) {
  (void)in_sizes; (void)n_in; (void)out_size; (void)ws_size;
  const float* feat   = (const float*)d_in[0];
  const float* labels = (const float*)d_in[1];
  float* out = (float*)d_out;

  char* ws = (char*)d_ws;
  _Float16* Abuf     = (_Float16*)(ws);                               // 2 MB
  _Float16* Bbuf     = (_Float16*)(ws + (size_t)(2u << 20));          // 2 MB
  unsigned* maskbits = (unsigned*)(ws + (size_t)(4u << 20));          // 512 KB
  float*    accum    = (float*)(ws + (size_t)(4u << 20) + (512u << 10));

  prep_kernel<<<NN, EDIM, 0, stream>>>(feat, Abuf, Bbuf);
  mask_kernel<<<(BSZ * 64 + 255) / 256, 256, 0, stream>>>(labels, maskbits);
  zero_kernel<<<1, 1, 0, stream>>>(accum);
  scl_main_kernel<<<NTI, 128, 0, stream>>>(Abuf, Bbuf, maskbits, accum);
  fin_kernel<<<1, 1, 0, stream>>>(accum, out);
}
